// SpatialAttention_75015898792409
// MI455X (gfx1250) — matrix operand order repaired and checked by reading, not compiled, not run on hardware
//
#include <hip/hip_runtime.h>
#include <math.h>

#define DM 512
#define HN 8
#define HD 64
#define NT 2048
#define NB 2
#define NS 32
#define MROWS (NB*NT)   // 4096
#define DHID 128

typedef __attribute__((ext_vector_type(16))) _Float16 v16h;
typedef __attribute__((ext_vector_type(8)))  float    v8f;

union AFrag { v16h v; uint4 q[2]; };

__device__ inline v8f v8f_zero() {
  v8f z = {0.f,0.f,0.f,0.f,0.f,0.f,0.f,0.f};
  return z;
}

// A-matrix fragment 16x32 f16: lane (hl,hi): row = row0+hl,
// elems 0..7 -> K = hi*8+0..7 ; elems 8..15 -> K = 16+hi*8+0..7
__device__ inline void load_fragA(AFrag& f, const _Float16* base, int ld) {
  const int lane = threadIdx.x & 31;
  const int hl = lane & 15, hi = lane >> 4;
  const _Float16* p = base + hl * ld + hi * 8;
  f.q[0] = *reinterpret_cast<const uint4*>(p);
  f.q[1] = *reinterpret_cast<const uint4*>(p + 16);
}

// B-matrix fragment 32x16 f16, loaded from transposed storage Bt[n][k]:
// lane (hl,hi): col n = n0+hl, elems 0..7 -> K = hi*8+0..7 ; elems 8..15 -> K = 16+hi*8+0..7
__device__ inline void load_fragB(AFrag& f, const _Float16* base, int ld) {
  const int lane = threadIdx.x & 31;
  const int hl = lane & 15, hi = lane >> 4;
  const _Float16* p = base + hl * ld + hi * 8;
  f.q[0] = *reinterpret_cast<const uint4*>(p);
  f.q[1] = *reinterpret_cast<const uint4*>(p + 16);
}

__device__ inline v8f wmma16(const AFrag& a, const AFrag& b, v8f c) {
  return __builtin_amdgcn_wmma_f32_16x16x32_f16(false, a.v, false, b.v,
                                                (short)0, c, false, false);
}

// ---------------- conversion kernels ----------------
__global__ void cvt_f32_f16(const float* __restrict__ in, _Float16* __restrict__ out, int n) {
  int i = blockIdx.x * blockDim.x + threadIdx.x;
  if (i < n) out[i] = (_Float16)in[i];
}

__global__ void transpose_f32_f16(const float* __restrict__ W, _Float16* __restrict__ Wt) {
  int i = blockIdx.x * blockDim.x + threadIdx.x;   // DM*DM
  int r = i >> 9, c = i & (DM - 1);
  Wt[c * DM + r] = (_Float16)W[r * DM + c];
}

// ---------------- distance MLP bias ----------------
__global__ void dist_bias_kernel(const float* __restrict__ coords,
                                 const float* __restrict__ Wd1, const float* __restrict__ bd1,
                                 const float* __restrict__ Wd2, const float* __restrict__ bd2,
                                 float* __restrict__ bias) {
  int id = blockIdx.x * blockDim.x + threadIdx.x;
  if (id >= NB * NS * NS) return;
  int b = id / (NS * NS);
  int i = (id / NS) % NS;
  int j = id % NS;
  const float R = 0.017453292519943295f;
  float lati = coords[(b * NS + i) * 2 + 0] * R, loni = coords[(b * NS + i) * 2 + 1] * R;
  float latj = coords[(b * NS + j) * 2 + 0] * R, lonj = coords[(b * NS + j) * 2 + 1] * R;
  float sa = sinf((latj - lati) * 0.5f);
  float sb = sinf((lonj - loni) * 0.5f);
  float a = sa * sa + cosf(lati) * cosf(latj) * sb * sb;
  float c = 2.0f * atan2f(sqrtf(a), sqrtf(fmaxf(1.0f - a, 0.0f)));
  float dist = 6371.0f * c;
  float acc[HN];
#pragma unroll
  for (int hh = 0; hh < HN; ++hh) acc[hh] = bd2[hh];
  for (int k2 = 0; k2 < DHID; ++k2) {
    float hv = fmaxf(dist * Wd1[k2] + bd1[k2], 0.0f);
#pragma unroll
    for (int hh = 0; hh < HN; ++hh) acc[hh] += hv * Wd2[k2 * HN + hh];
  }
#pragma unroll
  for (int hh = 0; hh < HN; ++hh)
    bias[(((b * HN + hh) * NS) + i) * NS + j] = acc[hh];
}

// ---------------- QKV projection GEMM (64x64 tile / wave) ----------------
__global__ void __launch_bounds__(32, 1)
qkv_gemm(const _Float16* __restrict__ xh,
         const _Float16* __restrict__ Wqt, const _Float16* __restrict__ Wkt,
         const _Float16* __restrict__ Wvt,
         const float* __restrict__ bq, const float* __restrict__ bk,
         const float* __restrict__ bv,
         _Float16* __restrict__ Qo, _Float16* __restrict__ Ko,
         _Float16* __restrict__ Vto) {
  const int row0 = blockIdx.x * 64;
  const int col0 = blockIdx.y * 64;
  const int sel = blockIdx.z;
  const _Float16* Wt = (sel == 0) ? Wqt : ((sel == 1) ? Wkt : Wvt);
  const float* bb = (sel == 0) ? bq : ((sel == 1) ? bk : bv);

  v8f acc[4][4];
#pragma unroll
  for (int i = 0; i < 4; ++i)
#pragma unroll
    for (int j = 0; j < 4; ++j) acc[i][j] = v8f_zero();

  for (int kk = 0; kk < DM; kk += 32) {
    AFrag a[4], b[4];
#pragma unroll
    for (int i = 0; i < 4; ++i) load_fragA(a[i], xh + (row0 + 16 * i) * DM + kk, DM);
#pragma unroll
    for (int j = 0; j < 4; ++j) load_fragB(b[j], Wt + (col0 + 16 * j) * DM + kk, DM);
#pragma unroll
    for (int i = 0; i < 4; ++i)
#pragma unroll
      for (int j = 0; j < 4; ++j) acc[i][j] = wmma16(a[i], b[j], acc[i][j]);
  }

  const int lane = threadIdx.x & 31, hl = lane & 15, hi = lane >> 4;
#pragma unroll
  for (int j = 0; j < 4; ++j) {
    int col = col0 + 16 * j + hl;
    float bval = bb[col];
    int hh = col >> 6, d = col & 63;
#pragma unroll
    for (int i = 0; i < 4; ++i) {
#pragma unroll
      for (int e = 0; e < 8; ++e) {
        int row = row0 + 16 * i + 8 * hi + e;
        int bidx = row >> 11, t = row & (NT - 1);
        float vv = acc[i][j][e] + bval;
        if (sel == 0)
          Qo[(((size_t)(bidx * HN + hh)) * NT + t) * HD + d] = (_Float16)vv;
        else if (sel == 1)
          Ko[(((size_t)(bidx * HN + hh)) * NT + t) * HD + d] = (_Float16)vv;
        else
          Vto[(((size_t)(bidx * HN + hh)) * HD + d) * NT + t] = (_Float16)vv;
      }
    }
  }
}

// ---------------- fused flash attention: 32 q-rows per wave ----------------
__global__ void __launch_bounds__(32, 1)
flash_attn(const _Float16* __restrict__ Q, const _Float16* __restrict__ K,
           const _Float16* __restrict__ Vt, const float* __restrict__ bias,
           const unsigned char* __restrict__ smask,
           _Float16* __restrict__ attn) {
  __shared__ _Float16 Pl[32 * 72];   // padded stride 72 halves -> conflict-free frag reads
  const int q0 = blockIdx.x * 32;
  const int h = blockIdx.y;
  const int b = blockIdx.z;
  const int qs = q0 >> 6;            // q-station (T=64, q-tile=32 within one station)
  const _Float16* Qb = Q + ((size_t)(b * HN + h)) * NT * HD;
  const _Float16* Kb = K + ((size_t)(b * HN + h)) * NT * HD;
  const _Float16* Vb = Vt + ((size_t)(b * HN + h)) * HD * NT;
  const int lane = threadIdx.x & 31, hl = lane & 15, hi = lane >> 4;

  AFrag qf[2][2];
#pragma unroll
  for (int i = 0; i < 2; ++i)
#pragma unroll
    for (int kk2 = 0; kk2 < 2; ++kk2)
      load_fragA(qf[i][kk2], Qb + (q0 + 16 * i) * HD + kk2 * 32, HD);

  v8f o[2][4];
#pragma unroll
  for (int i = 0; i < 2; ++i)
#pragma unroll
    for (int nd = 0; nd < 4; ++nd) o[i][nd] = v8f_zero();

  float mrow[16], lrow[16];
#pragma unroll
  for (int idx = 0; idx < 16; ++idx) { mrow[idx] = -INFINITY; lrow[idx] = 0.f; }

  for (int kt = 0; kt < NS; ++kt) {          // one k-station (64 keys) per step
    if (!smask[b * NS + kt]) continue;       // masked station == -inf scores
    const float bsc = bias[(((b * HN + h) * NS) + qs) * NS + kt];

    v8f s[2][4];
#pragma unroll
    for (int i = 0; i < 2; ++i)
#pragma unroll
      for (int n = 0; n < 4; ++n) s[i][n] = v8f_zero();

#pragma unroll
    for (int n = 0; n < 4; ++n)
#pragma unroll
      for (int kk2 = 0; kk2 < 2; ++kk2) {
        AFrag kf;   // K row-major == B-fragment layout for K^T
        load_fragB(kf, Kb + (kt * 64 + 16 * n) * HD + kk2 * 32, HD);
#pragma unroll
        for (int i = 0; i < 2; ++i) s[i][n] = wmma16(qf[i][kk2], kf, s[i][n]);
      }

#pragma unroll
    for (int i = 0; i < 2; ++i)
#pragma unroll
      for (int n = 0; n < 4; ++n)
#pragma unroll
        for (int e = 0; e < 8; ++e) s[i][n][e] = s[i][n][e] * 0.125f + bsc;

    // online softmax: rows of a (i,e) slot live in 16 lanes of the same half
#pragma unroll
    for (int i = 0; i < 2; ++i)
#pragma unroll
      for (int e = 0; e < 8; ++e) {
        const int idx = i * 8 + e;
        float mx = fmaxf(fmaxf(s[i][0][e], s[i][1][e]), fmaxf(s[i][2][e], s[i][3][e]));
#pragma unroll
        for (int m = 1; m < 16; m <<= 1) mx = fmaxf(mx, __shfl_xor(mx, m, 32));
        float mnew = fmaxf(mrow[idx], mx);
        float alpha = __expf(mrow[idx] - mnew);
        mrow[idx] = mnew;
        float rs = 0.f;
#pragma unroll
        for (int n = 0; n < 4; ++n) {
          float p = __expf(s[i][n][e] - mnew);
          s[i][n][e] = p;
          rs += p;
        }
#pragma unroll
        for (int m = 1; m < 16; m <<= 1) rs += __shfl_xor(rs, m, 32);
        lrow[idx] = lrow[idx] * alpha + rs;
#pragma unroll
        for (int nd = 0; nd < 4; ++nd) o[i][nd][e] *= alpha;
      }

    // re-layout P (C-layout -> A-layout) through LDS
    __syncthreads();
#pragma unroll
    for (int i = 0; i < 2; ++i)
#pragma unroll
      for (int n = 0; n < 4; ++n)
#pragma unroll
        for (int e = 0; e < 8; ++e)
          Pl[(16 * i + 8 * hi + e) * 72 + 16 * n + hl] = (_Float16)s[i][n][e];
    __syncthreads();

#pragma unroll
    for (int kk2 = 0; kk2 < 2; ++kk2) {
      AFrag pf[2];
      load_fragA(pf[0], &Pl[0 * 72 + kk2 * 32], 72);
      load_fragA(pf[1], &Pl[16 * 72 + kk2 * 32], 72);
#pragma unroll
      for (int nd = 0; nd < 4; ++nd) {
        AFrag vf;   // V^T row-major == B-fragment layout for V
        load_fragB(vf, Vb + (16 * nd) * NT + kt * 64 + kk2 * 32, NT);
#pragma unroll
        for (int i = 0; i < 2; ++i) o[i][nd] = wmma16(pf[i], vf, o[i][nd]);
      }
    }
  }

  // normalize and write (B, NT, D) f16 for the output projection
#pragma unroll
  for (int i = 0; i < 2; ++i)
#pragma unroll
    for (int e = 0; e < 8; ++e) {
      const int idx = i * 8 + e;
      float inv = 1.0f / lrow[idx];
      int t = q0 + 16 * i + 8 * hi + e;
#pragma unroll
      for (int nd = 0; nd < 4; ++nd)
        attn[((size_t)(b * NT + t)) * DM + h * HD + 16 * nd + hl] =
            (_Float16)(o[i][nd][e] * inv);
    }
}

// ---------------- output projection GEMM -> f32 ----------------
__global__ void __launch_bounds__(32, 1)
out_gemm(const _Float16* __restrict__ ah, const _Float16* __restrict__ Wot,
         const float* __restrict__ bo, float* __restrict__ out) {
  const int row0 = blockIdx.x * 64;
  const int col0 = blockIdx.y * 64;
  v8f acc[4][4];
#pragma unroll
  for (int i = 0; i < 4; ++i)
#pragma unroll
    for (int j = 0; j < 4; ++j) acc[i][j] = v8f_zero();

  for (int kk = 0; kk < DM; kk += 32) {
    AFrag a[4], b[4];
#pragma unroll
    for (int i = 0; i < 4; ++i) load_fragA(a[i], ah + (row0 + 16 * i) * DM + kk, DM);
#pragma unroll
    for (int j = 0; j < 4; ++j) load_fragB(b[j], Wot + (col0 + 16 * j) * DM + kk, DM);
#pragma unroll
    for (int i = 0; i < 4; ++i)
#pragma unroll
      for (int j = 0; j < 4; ++j) acc[i][j] = wmma16(a[i], b[j], acc[i][j]);
  }

  const int lane = threadIdx.x & 31, hl = lane & 15, hi = lane >> 4;
#pragma unroll
  for (int j = 0; j < 4; ++j) {
    int col = col0 + 16 * j + hl;
    float bval = bo[col];
#pragma unroll
    for (int i = 0; i < 4; ++i)
#pragma unroll
      for (int e = 0; e < 8; ++e) {
        int row = row0 + 16 * i + 8 * hi + e;
        out[(size_t)row * DM + col] = acc[i][j][e] + bval;
      }
  }
}

extern "C" void kernel_launch(void* const* d_in, const int* in_sizes, int n_in,
                              void* d_out, int out_size, void* d_ws, size_t ws_size,
                              hipStream_t stream) {
  const float* x = (const float*)d_in[0];
  const float* coords = (const float*)d_in[1];
  const unsigned char* smask = (const unsigned char*)d_in[2];
  const float* Wq = (const float*)d_in[3];
  const float* bq = (const float*)d_in[4];
  const float* Wk = (const float*)d_in[5];
  const float* bk = (const float*)d_in[6];
  const float* Wv = (const float*)d_in[7];
  const float* bv = (const float*)d_in[8];
  const float* Wo = (const float*)d_in[9];
  const float* bo = (const float*)d_in[10];
  const float* Wd1 = (const float*)d_in[11];
  const float* bd1 = (const float*)d_in[12];
  const float* Wd2 = (const float*)d_in[13];
  const float* bd2 = (const float*)d_in[14];
  float* out = (float*)d_out;

  char* ws = (char*)d_ws;
  size_t off = 0;
  auto alloc = [&](size_t bytes) -> void* {
    void* p = ws + off;
    off += (bytes + 255) & ~(size_t)255;
    return p;
  };
  _Float16* xh  = (_Float16*)alloc((size_t)MROWS * DM * 2);
  _Float16* Wqt = (_Float16*)alloc((size_t)DM * DM * 2);
  _Float16* Wkt = (_Float16*)alloc((size_t)DM * DM * 2);
  _Float16* Wvt = (_Float16*)alloc((size_t)DM * DM * 2);
  _Float16* Wot = (_Float16*)alloc((size_t)DM * DM * 2);
  _Float16* Qh  = (_Float16*)alloc((size_t)NB * HN * NT * HD * 2);
  _Float16* Kh  = (_Float16*)alloc((size_t)NB * HN * NT * HD * 2);
  _Float16* Vth = (_Float16*)alloc((size_t)NB * HN * HD * NT * 2);
  _Float16* ah  = (_Float16*)alloc((size_t)MROWS * DM * 2);
  float*    bias = (float*)alloc((size_t)NB * HN * NS * NS * 4);

  cvt_f32_f16<<<(MROWS * DM + 255) / 256, 256, 0, stream>>>(x, xh, MROWS * DM);
  transpose_f32_f16<<<(DM * DM + 255) / 256, 256, 0, stream>>>(Wq, Wqt);
  transpose_f32_f16<<<(DM * DM + 255) / 256, 256, 0, stream>>>(Wk, Wkt);
  transpose_f32_f16<<<(DM * DM + 255) / 256, 256, 0, stream>>>(Wv, Wvt);
  transpose_f32_f16<<<(DM * DM + 255) / 256, 256, 0, stream>>>(Wo, Wot);
  dist_bias_kernel<<<(NB * NS * NS + 127) / 128, 128, 0, stream>>>(coords, Wd1, bd1, Wd2, bd2, bias);

  qkv_gemm<<<dim3(MROWS / 64, DM / 64, 3), 32, 0, stream>>>(xh, Wqt, Wkt, Wvt, bq, bk, bv,
                                                            Qh, Kh, Vth);
  flash_attn<<<dim3(NT / 32, HN, NB), 32, 0, stream>>>(Qh, Kh, Vth, bias, smask, ah);
  out_gemm<<<dim3(MROWS / 64, DM / 64), 32, 0, stream>>>(ah, Wot, bo, out);
}
